// IntraVoxelVolterra_8375186227640
// MI455X (gfx1250) — matrix operand order repaired and checked by reading, not compiled, not run on hardware
//
#include <hip/hip_runtime.h>
#include <hip/hip_bf16.h>

typedef __attribute__((ext_vector_type(16))) __bf16 v16bf;
typedef __attribute__((ext_vector_type(8)))  __bf16 v8bf;
typedef __attribute__((ext_vector_type(8)))  float  v8f;

#define D 64
#define CI 32
#define CO 32
#define TAPS 27            // 3*3*3
#define XT 32              // x-tile (block M-tile)
#define PATCH_X 34         // XT + 2 halo
#define PATCH_VOX (9 * PATCH_X)      // 3*3*34 = 306
#define PATCH_ELEMS (PATCH_VOX * CI) // 9792
#define FRAG_HALVES 512    // 32 lanes * 16 halves per B fragment
#define CONV_STRIDE (TAPS * 2 * FRAG_HALVES)  // 27648 halves per conv in ws

__device__ __forceinline__ unsigned short f2bf(float f) {
    unsigned u = __float_as_uint(f);
    unsigned r = u + 0x7FFF + ((u >> 16) & 1);   // round-to-nearest-even
    return (unsigned short)(r >> 16);
}

// ---------------------------------------------------------------------------
// Pack w1/w2/w3 (fp32, [CO,3,3,3,CI]) into bf16 B-fragments in ws, in the
// per-lane B layout this file uses for V_WMMA_F32_16X16X32_BF16 (K=ci, N=co):
//   lane l (0..15):  co = ntile*16 + l,      halves h -> ci = h
//   lane l (16..31): co = ntile*16 + (l-16), halves h -> ci = 16 + h
// Fragment order: [conv c][tap t][ntile n][lane][half]
// ---------------------------------------------------------------------------
__global__ void volterra_pack_w(const float* __restrict__ w1,
                                const float* __restrict__ w2,
                                const float* __restrict__ w3,
                                unsigned short* __restrict__ ws) {
    int idx = blockIdx.x * 256 + threadIdx.x;
    const int total = 3 * CONV_STRIDE;           // 82944
    if (idx >= total) return;
    int c    = idx / CONV_STRIDE;
    int rem  = idx - c * CONV_STRIDE;
    int t    = rem / (2 * FRAG_HALVES);
    int rem2 = rem - t * (2 * FRAG_HALVES);
    int n    = rem2 / FRAG_HALVES;
    int q    = rem2 - n * FRAG_HALVES;
    int l    = q >> 4;
    int h    = q & 15;
    int co   = n * 16 + (l & 15);
    int ci   = h + ((l >= 16) ? 16 : 0);
    const float* w = (c == 0) ? w1 : ((c == 1) ? w2 : w3);
    ws[idx] = f2bf(w[co * (TAPS * CI) + t * CI + ci]);
}

// ---------------------------------------------------------------------------
// Main fused kernel: stage log(x) patch (bf16) in LDS, implicit-GEMM via
// v_wmma_f32_16x16x32_bf16. 64 threads = 2 waves; wave wid owns N-tile wid
// and the FULL M=32 (two 16-row tiles), so each B fragment feeds 2 WMMAs:
// 27 taps x 3 convs x 2 m-tiles = 162 WMMAs per wave.
// Grid: B*D*D*(D/XT) = 16384 blocks.
// ---------------------------------------------------------------------------
__global__ void __launch_bounds__(64)
volterra_main(const float* __restrict__ x,
              const unsigned short* __restrict__ ws,
              float* __restrict__ out) {
    __shared__ unsigned short sA[PATCH_ELEMS];   // [vox][ci] bf16, 19.6 KB

    const int bid   = blockIdx.x;
    const int xtile = bid & 1;
    const int yy    = (bid >> 1) & 63;
    const int zz    = (bid >> 7) & 63;
    const int b     = bid >> 13;
    const int tid   = threadIdx.x;

    // ---- Stage: load 3x3x34 halo patch, log(), cvt bf16, store LDS --------
    for (int e = tid; e < PATCH_ELEMS; e += 64) {
        int ci  = e / PATCH_VOX;
        int v   = e - ci * PATCH_VOX;
        int row = v / PATCH_X;          // dz*3+dy
        int xi  = v - row * PATCH_X;    // 0..33
        int dz  = row / 3;
        int dy  = row - dz * 3;
        int z2  = zz + dz - 1;
        int y2  = yy + dy - 1;
        int x2  = xtile * XT + xi - 1;
        float val = 0.0f;               // zero-pad in log domain (factor 1.0)
        if ((unsigned)z2 < D && (unsigned)y2 < D && (unsigned)x2 < D) {
            val = __logf(x[(((b * CI + ci) * D + z2) * D + y2) * D + x2]);
        }
        sA[v * CI + ci] = f2bf(val);
    }
    __syncthreads();

    // ---- Compute: 2 waves; ni = wave id; each wave does both m-tiles ------
    const int lane = tid & 31;
    const int ni   = tid >> 5;             // N-tile (0/1)
    const int r    = lane & 15;
    const int hi   = lane >> 4;            // lane half (A/B K-split)

    v8f acc1a = {}, acc1b = {};
    v8f acc2a = {}, acc2b = {};
    v8f acc3a = {}, acc3b = {};

    const unsigned short* fragBase = ws + ni * FRAG_HALVES + lane * 16;

    for (int dz = 0; dz < 3; ++dz) {
        for (int dy = 0; dy < 3; ++dy) {
            const int rowBase = (dz * 3 + dy) * PATCH_X;
#pragma unroll
            for (int dx = 0; dx < 3; ++dx) {
                // A fragments for m-tile 0 (rows 0..15) and 1 (rows 16..31):
                // lane needs ci [hi*16..+15] (matches B packing)
                // -> two contiguous 16B LDS chunks each.
                int vox0 = rowBase + r + dx;
                const unsigned short* ap0 = &sA[vox0 * CI + hi * 16];
                v8bf a0lo = *(const v8bf*)ap0;
                v8bf a0hi = *(const v8bf*)(ap0 + 8);
                v16bf a0 = __builtin_shufflevector(a0lo, a0hi,
                        0, 1, 2, 3, 4, 5, 6, 7, 8, 9, 10, 11, 12, 13, 14, 15);
                const unsigned short* ap1 = ap0 + 16 * CI;
                v8bf a1lo = *(const v8bf*)ap1;
                v8bf a1hi = *(const v8bf*)(ap1 + 8);
                v16bf a1 = __builtin_shufflevector(a1lo, a1hi,
                        0, 1, 2, 3, 4, 5, 6, 7, 8, 9, 10, 11, 12, 13, 14, 15);

                int t = (dz * 3 + dy) * 3 + dx;
                const unsigned short* bp = fragBase + t * (2 * FRAG_HALVES);
                v16bf b1 = *(const v16bf*)(bp);
                v16bf b2 = *(const v16bf*)(bp + CONV_STRIDE);
                v16bf b3 = *(const v16bf*)(bp + 2 * CONV_STRIDE);

                acc1a = __builtin_amdgcn_wmma_f32_16x16x32_bf16(
                            false, a0, false, b1, (short)0, acc1a, false, false);
                acc1b = __builtin_amdgcn_wmma_f32_16x16x32_bf16(
                            false, a1, false, b1, (short)0, acc1b, false, false);
                acc2a = __builtin_amdgcn_wmma_f32_16x16x32_bf16(
                            false, a0, false, b2, (short)0, acc2a, false, false);
                acc2b = __builtin_amdgcn_wmma_f32_16x16x32_bf16(
                            false, a1, false, b2, (short)0, acc2b, false, false);
                acc3a = __builtin_amdgcn_wmma_f32_16x16x32_bf16(
                            false, a0, false, b3, (short)0, acc3a, false, false);
                acc3b = __builtin_amdgcn_wmma_f32_16x16x32_bf16(
                            false, a1, false, b3, (short)0, acc3b, false, false);
            }
        }
    }

    // ---- Epilogue: out = x * exp(acc1 + acc2*acc3) ------------------------
    // f32 C/D layout: VGPR g, lanes 0-15 -> M=g, lanes 16-31 -> M=g+8; N=lane&15
    const int co   = ni * 16 + (lane & 15);
    const int mOff = hi * 8;
    const int base = (((b * CO + co) * D + zz) * D + yy) * D + xtile * XT;
#pragma unroll
    for (int g = 0; g < 8; ++g) {
        int idx0 = base + mOff + g;              // m-tile 0
        float xv0 = x[idx0];
        out[idx0] = xv0 * __expf(acc1a[g] + acc2a[g] * acc3a[g]);
        int idx1 = idx0 + 16;                    // m-tile 1
        float xv1 = x[idx1];
        out[idx1] = xv1 * __expf(acc1b[g] + acc2b[g] * acc3b[g]);
    }
}

extern "C" void kernel_launch(void* const* d_in, const int* in_sizes, int n_in,
                              void* d_out, int out_size, void* d_ws, size_t ws_size,
                              hipStream_t stream) {
    const float* x  = (const float*)d_in[0];
    const float* w1 = (const float*)d_in[1];
    const float* w2 = (const float*)d_in[2];
    const float* w3 = (const float*)d_in[3];
    unsigned short* wsb = (unsigned short*)d_ws;   // 3*27648 bf16 = 162 KB
    float* out = (float*)d_out;

    // Pack weights into WMMA B-fragment layout (bf16) in workspace.
    volterra_pack_w<<<(3 * CONV_STRIDE + 255) / 256, 256, 0, stream>>>(w1, w2, w3, wsb);

    // Fused log -> 3x implicit-GEMM conv (WMMA bf16) -> exp epilogue.
    const int blocks = 2 * D * D * (D / XT);       // 16384
    volterra_main<<<blocks, 64, 0, stream>>>(x, wsb, out);
}
